// AttnBlock_11278584119532
// MI455X (gfx1250) — matrix operand order repaired and checked by reading, not compiled, not run on hardware
//
#include <hip/hip_runtime.h>
#include <hip/hip_bf16.h>
#include <math.h>

typedef __attribute__((ext_vector_type(16))) __bf16 v16bf;
typedef __attribute__((ext_vector_type(8)))  __bf16 v8bf;
typedef __attribute__((ext_vector_type(8)))  float  v8f;

#define C_DIM 512
#define N_DIM 4096
#define B_DIM 4
#define GROUPS 32
#define RSCALE 0.04419417382415922f   // 512^-0.5

__device__ __forceinline__ v8f wmma_bf16(v16bf a, v16bf b, v8f c) {
  // D = A(16x32 bf16) * B(32x16 bf16) + C(16x16 f32)
  return __builtin_amdgcn_wmma_f32_16x16x32_bf16(false, a, false, b, (short)0, c,
                                                 false, false);
}

// A-operand layout: lane L(0-15) row M=L holds K = [c0..c0+15];
// lane L+16 same row holds K = [c0+16..c0+31].
// Caller passes off0 = c0 + (lane>=16 ? 16 : 0).
__device__ __forceinline__ v16bf load_a16(const __bf16* base, int off0) {
  v8bf lo = *(const v8bf*)(base + off0);
  v8bf hi = *(const v8bf*)(base + off0 + 8);
  v16bf r;
#pragma unroll
  for (int e = 0; e < 8; ++e) { r[e] = lo[e]; r[8 + e] = hi[e]; }
  return r;
}

__device__ __forceinline__ v8f zero8() {
  v8f z;
#pragma unroll
  for (int e = 0; e < 8; ++e) z[e] = 0.0f;
  return z;
}

// ---------------- GroupNorm: x(b,c,n) f32 -> h_t(b,n,c) bf16 ----------------
__global__ void gn_kernel(const float* __restrict__ x,
                          const float* __restrict__ sc,
                          const float* __restrict__ bi,
                          __bf16* __restrict__ h_t) {
  int g = blockIdx.x, b = blockIdx.y, tid = threadIdx.x;
  const int CG = C_DIM / GROUPS;             // 16 channels per group
  const int ELE = CG * N_DIM;                // 65536
  const float* xg = x + ((size_t)b * C_DIM + g * CG) * N_DIM;
  float s = 0.f, ss = 0.f;
  for (int idx = tid; idx < ELE; idx += 256) {
    float v = xg[idx];
    s += v; ss += v * v;
  }
  __shared__ float rs[256], rss[256];
  rs[tid] = s; rss[tid] = ss;
  __syncthreads();
  for (int off = 128; off > 0; off >>= 1) {
    if (tid < off) { rs[tid] += rs[tid + off]; rss[tid] += rss[tid + off]; }
    __syncthreads();
  }
  float mean = rs[0] / (float)ELE;
  float var  = rss[0] / (float)ELE - mean * mean;
  float rstd = rsqrtf(var + 1e-6f);
  for (int idx = tid; idx < ELE; idx += 256) {
    int cl = idx >> 12;
    int i  = idx & (N_DIM - 1);
    int c  = g * CG + cl;
    float v = (xg[idx] - mean) * rstd * sc[c] + bi[c];
    h_t[((size_t)b * N_DIM + i) * C_DIM + c] = (__bf16)v;
  }
}

// ---------------- weight f32 -> bf16 ----------------
__global__ void cvt_kernel(const float* __restrict__ w, __bf16* __restrict__ o, int n) {
  int i = blockIdx.x * 256 + threadIdx.x;
  if (i < n) o[i] = (__bf16)w[i];
}

// ---------------- QKV GEMMs: out[d,i] = W[d,:]·h_t[i,:] + bias[d] ----------------
// q,k stored (b,n,c) bf16; v stored (b,c,n) bf16.
__global__ void qkv_kernel(const __bf16* __restrict__ h_t,
                           const __bf16* __restrict__ wqb,
                           const __bf16* __restrict__ wkb,
                           const __bf16* __restrict__ wvb,
                           const float* __restrict__ bq,
                           const float* __restrict__ bk,
                           const float* __restrict__ bv,
                           __bf16* __restrict__ q_t,
                           __bf16* __restrict__ k_t,
                           __bf16* __restrict__ vbuf) {
  int wave = threadIdx.x >> 5, lane = threadIdx.x & 31;
  int nn = lane & 15, hi = lane >> 4, hi8 = hi * 8, hi16 = hi * 16;
  int b = blockIdx.z / 3, which = blockIdx.z % 3;
  int i0 = blockIdx.x * 16;
  int d0 = (blockIdx.y * 4 + wave) * 16;
  const __bf16* W = (which == 0) ? wqb : (which == 1) ? wkb : wvb;
  const float* bias = (which == 0) ? bq : (which == 1) ? bk : bv;
  v8f acc;
#pragma unroll
  for (int r = 0; r < 8; ++r) acc[r] = bias[d0 + hi8 + r];
  const __bf16* arow = W + (size_t)(d0 + nn) * C_DIM;
  const __bf16* brow = h_t + ((size_t)b * N_DIM + i0 + nn) * C_DIM + hi16;
#pragma unroll
  for (int c0 = 0; c0 < C_DIM; c0 += 32) {
    v16bf A = load_a16(arow, c0 + hi16);
    v16bf B = *(const v16bf*)(brow + c0);
    acc = wmma_bf16(A, B, acc);
  }
  if (which < 2) {
    // D tile: lane holds rows d0+hi8..d0+hi8+7 at fixed i = i0+nn -> contiguous 16B
    __bf16* out = ((which == 0) ? q_t : k_t) +
                  ((size_t)b * N_DIM + i0 + nn) * C_DIM + d0 + hi8;
    v8bf pk;
#pragma unroll
    for (int r = 0; r < 8; ++r) pk[r] = (__bf16)acc[r];
    *(v8bf*)out = pk;
  } else {
#pragma unroll
    for (int r = 0; r < 8; ++r)
      vbuf[((size_t)b * C_DIM + d0 + hi8 + r) * N_DIM + i0 + nn] = (__bf16)acc[r];
  }
}

// scores for a 16-query x 32-key chunk: S = q_tile^T applied as A, k rows as B
__device__ __forceinline__ void score_chunk(const __bf16* qrow, const __bf16* kb,
                                            int j0, int nn, int hi8, int hi16,
                                            v8f& S0, v8f& S1) {
  S0 = zero8(); S1 = zero8();
  const __bf16* k0 = kb + (size_t)(j0 + nn) * C_DIM + hi16;
  const __bf16* k1 = kb + (size_t)(j0 + 16 + nn) * C_DIM + hi16;
#pragma unroll
  for (int c0 = 0; c0 < C_DIM; c0 += 32) {
    v16bf A  = load_a16(qrow, c0 + hi16);
    v16bf B0 = *(const v16bf*)(k0 + c0);
    v16bf B1 = *(const v16bf*)(k1 + c0);
    S0 = wmma_bf16(A, B0, S0);
    S1 = wmma_bf16(A, B1, S1);
  }
}

// ---------------- attention: one block (4 waves) per 16-query tile ----------------
__global__ void attn_kernel(const __bf16* __restrict__ q_t,
                            const __bf16* __restrict__ k_t,
                            const __bf16* __restrict__ vbuf,
                            __bf16* __restrict__ attn_t) {
  __shared__ __align__(32) __bf16 psh[16][1024];   // prob tile for 1/4 of the keys
  __shared__ float msh[4][16];
  __shared__ float mfin[16];
  __shared__ float lsh[16];
  __shared__ float lpart[128];

  int tid = threadIdx.x;
  int wave = tid >> 5, lane = tid & 31;
  int nn = lane & 15, hi = lane >> 4, hi8 = hi * 8, hi16 = hi * 16;
  int i0 = blockIdx.x * 16, b = blockIdx.y;

  const __bf16* qb = q_t + (size_t)b * N_DIM * C_DIM;
  const __bf16* kb = k_t + (size_t)b * N_DIM * C_DIM;
  const __bf16* vb = vbuf + (size_t)b * C_DIM * N_DIM;
  const __bf16* qrow = qb + (size_t)(i0 + nn) * C_DIM;

  // ---- pass 0: row maxima over all 4096 keys ----
  float mrow[8];
#pragma unroll
  for (int r = 0; r < 8; ++r) mrow[r] = -3.0e38f;
  for (int jc = wave; jc < N_DIM / 32; jc += 4) {
    v8f S0, S1;
    score_chunk(qrow, kb, jc * 32, nn, hi8, hi16, S0, S1);
#pragma unroll
    for (int r = 0; r < 8; ++r) {
      float cm = fmaxf(S0[r], S1[r]) * RSCALE;
      cm = fmaxf(cm, __shfl_xor(cm, 1, 32));
      cm = fmaxf(cm, __shfl_xor(cm, 2, 32));
      cm = fmaxf(cm, __shfl_xor(cm, 4, 32));
      cm = fmaxf(cm, __shfl_xor(cm, 8, 32));
      mrow[r] = fmaxf(mrow[r], cm);
    }
  }
  if (lane == 0) {
#pragma unroll
    for (int r = 0; r < 8; ++r) msh[wave][r] = mrow[r];
  } else if (lane == 16) {
#pragma unroll
    for (int r = 0; r < 8; ++r) msh[wave][8 + r] = mrow[r];
  }
  __syncthreads();
  if (tid < 16) {
    float m = msh[0][tid];
    m = fmaxf(m, msh[1][tid]); m = fmaxf(m, msh[2][tid]); m = fmaxf(m, msh[3][tid]);
    mfin[tid] = m;
    lsh[tid] = 0.0f;
  }
  __syncthreads();

  float mreg[8];
#pragma unroll
  for (int r = 0; r < 8; ++r) mreg[r] = mfin[hi8 + r];

  v8f acc[8];
#pragma unroll
  for (int t = 0; t < 8; ++t) acc[t] = zero8();

  for (int qtr = 0; qtr < 4; ++qtr) {
    // recompute scores, exponentiate, stage P tile in LDS (bf16)
    for (int jc = wave; jc < 32; jc += 4) {
      int j0 = qtr * 1024 + jc * 32;
      v8f S0, S1;
      score_chunk(qrow, kb, j0, nn, hi8, hi16, S0, S1);
#pragma unroll
      for (int r = 0; r < 8; ++r) {
        int row = hi8 + r;
        psh[row][jc * 32 + nn]      = (__bf16)__expf(S0[r] * RSCALE - mreg[r]);
        psh[row][jc * 32 + 16 + nn] = (__bf16)__expf(S1[r] * RSCALE - mreg[r]);
      }
    }
    __syncthreads();
    // accumulate softmax denominators
    {
      int row = tid & 15, cb = (tid >> 4) * 128;
      float s = 0.f;
      for (int k = 0; k < 128; ++k) s += (float)psh[row][cb + k];
      lpart[tid] = s;
    }
    __syncthreads();
    if (tid < 16) {
      float s = 0.f;
#pragma unroll
      for (int k = 0; k < 8; ++k) s += lpart[tid + 16 * k];
      lsh[tid] += s;
    }
    // O += P * V ; each wave owns 8 c-tiles (128 channels)
    const __bf16* prow = &psh[nn][0];
#pragma unroll 1
    for (int ctl = 0; ctl < 8; ++ctl) {
      int c0 = (wave * 8 + ctl) * 16;
      const __bf16* vcol = vb + (size_t)(c0 + nn) * N_DIM + qtr * 1024 + hi16;
      for (int jc = 0; jc < 32; ++jc) {
        v16bf A = load_a16(prow, jc * 32 + hi16);
        v16bf B = *(const v16bf*)(vcol + jc * 32);
        acc[ctl] = wmma_bf16(A, B, acc[ctl]);
      }
    }
    __syncthreads();
  }

  float lreg[8];
#pragma unroll
  for (int r = 0; r < 8; ++r) lreg[r] = lsh[hi8 + r];
#pragma unroll
  for (int ctl = 0; ctl < 8; ++ctl) {
    int c0 = (wave * 8 + ctl) * 16;
#pragma unroll
    for (int r = 0; r < 8; ++r)
      attn_t[((size_t)b * N_DIM + i0 + hi8 + r) * C_DIM + c0 + nn] =
          (__bf16)(acc[ctl][r] / lreg[r]);
  }
}

// ---------------- proj GEMM + bias + residual ----------------
__global__ void proj_kernel(const __bf16* __restrict__ attn_t,
                            const __bf16* __restrict__ wpb,
                            const float* __restrict__ bp,
                            const float* __restrict__ x,
                            float* __restrict__ out) {
  int wave = threadIdx.x >> 5, lane = threadIdx.x & 31;
  int nn = lane & 15, hi = lane >> 4, hi8 = hi * 8, hi16 = hi * 16;
  int b = blockIdx.z;
  int i0 = blockIdx.x * 16;
  int d0 = (blockIdx.y * 4 + wave) * 16;
  v8f acc;
#pragma unroll
  for (int r = 0; r < 8; ++r) acc[r] = bp[d0 + hi8 + r];
  const __bf16* arow = wpb + (size_t)(d0 + nn) * C_DIM;
  const __bf16* brow = attn_t + ((size_t)b * N_DIM + i0 + nn) * C_DIM + hi16;
#pragma unroll
  for (int c0 = 0; c0 < C_DIM; c0 += 32) {
    v16bf A = load_a16(arow, c0 + hi16);
    v16bf B = *(const v16bf*)(brow + c0);
    acc = wmma_bf16(A, B, acc);
  }
#pragma unroll
  for (int r = 0; r < 8; ++r) {
    size_t o = ((size_t)b * C_DIM + d0 + hi8 + r) * N_DIM + i0 + nn;
    out[o] = x[o] + acc[r];
  }
}

extern "C" void kernel_launch(void* const* d_in, const int* in_sizes, int n_in,
                              void* d_out, int out_size, void* d_ws, size_t ws_size,
                              hipStream_t stream) {
  (void)in_sizes; (void)n_in; (void)out_size; (void)ws_size;
  const float* x   = (const float*)d_in[0];
  const float* gsc = (const float*)d_in[1];
  const float* gbi = (const float*)d_in[2];
  const float* wq  = (const float*)d_in[3];
  const float* bq  = (const float*)d_in[4];
  const float* wk  = (const float*)d_in[5];
  const float* bk  = (const float*)d_in[6];
  const float* wv  = (const float*)d_in[7];
  const float* bv  = (const float*)d_in[8];
  const float* wp  = (const float*)d_in[9];
  const float* bp  = (const float*)d_in[10];
  float* out = (float*)d_out;

  char* ws = (char*)d_ws;
  const size_t TENS = (size_t)B_DIM * N_DIM * C_DIM * sizeof(__bf16); // 16 MiB
  __bf16* h_t  = (__bf16*)(ws);
  __bf16* q_t  = (__bf16*)(ws + TENS);
  __bf16* k_t  = (__bf16*)(ws + 2 * TENS);
  __bf16* vbf  = (__bf16*)(ws + 3 * TENS);
  __bf16* attn = (__bf16*)(ws + 4 * TENS);
  const size_t WSZ = (size_t)C_DIM * C_DIM * sizeof(__bf16);          // 512 KiB
  __bf16* wqb = (__bf16*)(ws + 5 * TENS);
  __bf16* wkb = (__bf16*)(ws + 5 * TENS + WSZ);
  __bf16* wvb = (__bf16*)(ws + 5 * TENS + 2 * WSZ);
  __bf16* wpb = (__bf16*)(ws + 5 * TENS + 3 * WSZ);

  int wn = C_DIM * C_DIM;
  cvt_kernel<<<dim3((wn + 255) / 256), 256, 0, stream>>>(wq, wqb, wn);
  cvt_kernel<<<dim3((wn + 255) / 256), 256, 0, stream>>>(wk, wkb, wn);
  cvt_kernel<<<dim3((wn + 255) / 256), 256, 0, stream>>>(wv, wvb, wn);
  cvt_kernel<<<dim3((wn + 255) / 256), 256, 0, stream>>>(wp, wpb, wn);
  gn_kernel<<<dim3(GROUPS, B_DIM), 256, 0, stream>>>(x, gsc, gbi, h_t);
  qkv_kernel<<<dim3(N_DIM / 16, 8, 3 * B_DIM), 128, 0, stream>>>(
      h_t, wqb, wkb, wvb, bq, bk, bv, q_t, k_t, vbf);
  attn_kernel<<<dim3(N_DIM / 16, B_DIM), 128, 0, stream>>>(q_t, k_t, vbf, attn);
  proj_kernel<<<dim3(N_DIM / 16, 8, B_DIM), 128, 0, stream>>>(attn, wpb, bp, x, out);
}
